// InvariantInteractionBlock_80805514707436
// MI455X (gfx1250) — matrix operand order repaired and checked by reading, not compiled, not run on hardware
//
#include <hip/hip_runtime.h>
#include <hip/hip_bf16.h>

// ---------------------------------------------------------------------------
// CDNA5 (gfx1250) fused invariant-interaction block.
// - All GEMMs on v_wmma_f32_16x16x32_f16 (wave32 WMMA)
// - Edge pipeline fully fused (no [E,128] intermediates in HBM)
// - Block weight staging via Tensor Data Mover (tensor_load_to_lds + TENSORcnt)
// - silu/sigmoid use hardware v_rcp_f32 (no IEEE divide expansion)
// - B-fragments batched ahead of WMMA chains to overlap ds loads
// ---------------------------------------------------------------------------

typedef __attribute__((ext_vector_type(16))) _Float16 v16h;
typedef __attribute__((ext_vector_type(8)))  _Float16 v8h;
typedef __attribute__((ext_vector_type(8)))  float    v8f;

#define D_DIM 128
#define PI_OVER_CUTOFF 0.6283185307179586f
#define CUTOFF_F 5.0f

#if defined(__has_builtin)
#if __has_builtin(__builtin_amdgcn_tensor_load_to_lds) && __has_builtin(__builtin_amdgcn_s_wait_tensorcnt)
#define HAVE_TDM 1
#endif
#endif

static __device__ __forceinline__ v8f wmma_f16(v16h a, v16h b, v8f c) {
  // 8 args: (neg_a, A, neg_b, B, c_mod, C, reuse_a, reuse_b)
  return __builtin_amdgcn_wmma_f32_16x16x32_f16(false, a, false, b, (short)0, c,
                                                false, false);
}
// Fast sigmoid/silu: v_exp_f32 + v_rcp_f32, no IEEE-divide expansion.
static __device__ __forceinline__ float sigmf(float x) {
  return __builtin_amdgcn_rcpf(1.0f + __expf(-x));
}
static __device__ __forceinline__ float siluf(float x) { return x * sigmf(x); }

// Pack 16 fp32 (two contiguous groups of 8) into a 16-half A-fragment register.
static __device__ __forceinline__ v16h pack16(float4 f0, float4 f1, float4 f2, float4 f3, float s) {
  v16h a;
  a[0]=(_Float16)(f0.x*s);  a[1]=(_Float16)(f0.y*s);  a[2]=(_Float16)(f0.z*s);  a[3]=(_Float16)(f0.w*s);
  a[4]=(_Float16)(f1.x*s);  a[5]=(_Float16)(f1.y*s);  a[6]=(_Float16)(f1.z*s);  a[7]=(_Float16)(f1.w*s);
  a[8]=(_Float16)(f2.x*s);  a[9]=(_Float16)(f2.y*s);  a[10]=(_Float16)(f2.z*s); a[11]=(_Float16)(f2.w*s);
  a[12]=(_Float16)(f3.x*s); a[13]=(_Float16)(f3.y*s); a[14]=(_Float16)(f3.z*s); a[15]=(_Float16)(f3.w*s);
  return a;
}

// Load one K-chunk A-fragment from per-wave row-major f16 staging [16][128].
// A 16x32 f16 layout: lanes 0-15 hold K {0..7, 16..23}; lanes 16-31 hold {8..15, 24..31}.
static __device__ __forceinline__ v16h load_a_stage(const _Float16* stage, int kc, int n16, int hi) {
  const _Float16* r = stage + n16 * D_DIM + kc * 32 + hi * 8;
  v8h lo = *(const v8h*)r;
  v8h hh = *(const v8h*)(r + 16);
  v16h a;
#pragma unroll
  for (int t = 0; t < 8; t++) { a[t] = lo[t]; a[8 + t] = hh[t]; }
  return a;
}

// Sum across the 16-lane half-groups (B/C layout column reduction).
static __device__ __forceinline__ float gsum16(float v) {
  v += __shfl_xor(v, 1, 32);
  v += __shfl_xor(v, 2, 32);
  v += __shfl_xor(v, 4, 32);
  v += __shfl_xor(v, 8, 32);
  return v;
}

#if HAVE_TDM
typedef unsigned int v4u_tdm __attribute__((ext_vector_type(4)));
typedef int v8i_tdm __attribute__((ext_vector_type(8)));
typedef int v4i_tdm __attribute__((ext_vector_type(4)));
// 2D TDM global->LDS load of (tile1 rows x tile0 dwords), contiguous
// (tensor_dim0_stride == tile0). Descriptor per CDNA5 ISA ch.8:
//  group0: [1:0]=count=1, [63:32]=lds_addr, [120:64]=global_addr, [127:126]=type=2
//  group1: data_size=4B, tensor_dim0/1, tile_dim0/1, tensor_dim0_stride
static __device__ __forceinline__ void tdm_load_2d_dwords(const void* gptr, unsigned lds_off,
                                                          unsigned tile0, unsigned tile1) {
  unsigned long long ga = (unsigned long long)gptr;
  v4u_tdm g0;
  g0.x = 1u;                                                // count = 1
  g0.y = lds_off;                                           // LDS byte address
  g0.z = (unsigned)(ga & 0xffffffffu);                      // global_addr[31:0]
  g0.w = (unsigned)((ga >> 32) & 0x1ffffffu) | (2u << 30);  // global_addr[56:32] | type=2
  v8i_tdm g1;
  g1[0] = 0x00020000;                                // data_size = 2 (4 bytes)
  g1[1] = (int)((tile0 & 0xffffu) << 16);            // tensor_dim0[15:0] @ bits 63:48
  g1[2] = (int)((tile0 >> 16) | (tile1 << 16));      // tensor_dim0[31:16] | tensor_dim1[15:0]
  g1[3] = (int)(tile0 << 16);                        // tile_dim0 @ bits 127:112
  g1[4] = (int)tile1;                                // tile_dim1 (tile_dim2 = 0 -> 2D)
  g1[5] = (int)tile0;                                // tensor_dim0_stride[31:0]
  g1[6] = 0;
  g1[7] = 0;
  v4i_tdm z4 = {0, 0, 0, 0};
  v8i_tdm z8 = {0, 0, 0, 0, 0, 0, 0, 0};
  // 6-arg toolchain form: (g0, g1, g2, g3, extra_group, cpol)
  __builtin_amdgcn_tensor_load_to_lds(g0, g1, z4, z4, z8, 0);
}
#endif

// ---------------------------------------------------------------------------
// Repack a row-major fp32 weight matrix W[K][128] into f16 WMMA B-fragments.
// Fragment (kc, c) covers K rows kc*32..+31 and cols c*16..+15. Lane order:
// lane = (n%16) + 16*hi; per-lane halves t: k_local = 8*hi + (t&7) + 16*(t>=8).
// ---------------------------------------------------------------------------
__global__ void pack_weights(const float* __restrict__ src, _Float16* __restrict__ dst, int Kdim) {
  int i = blockIdx.x * blockDim.x + threadIdx.x;
  int total = (Kdim / 32) * 8 * 512;
  if (i >= total) return;
  int frag = i >> 9;
  int rem  = i & 511;
  int lane = rem >> 4;
  int t    = rem & 15;
  int kc   = frag >> 3;
  int c    = frag & 7;
  int k = kc * 32 + ((t >> 3) << 4) + ((lane >> 4) << 3) + (t & 7);
  int n = c * 16 + (lane & 15);
  dst[i] = (_Float16)src[k * D_DIM + n];
}

// ---------------------------------------------------------------------------
// Pre-LayerNorm: one wave per row of x[N,128].
// ---------------------------------------------------------------------------
__global__ void ln_kernel(const float* __restrict__ x, const float* __restrict__ g,
                          const float* __restrict__ b, float* __restrict__ xn, int Nn) {
  int wave = (blockIdx.x * blockDim.x + threadIdx.x) >> 5;
  int lane = threadIdx.x & 31;
  if (wave >= Nn) return;
  const float* row = x + (size_t)wave * D_DIM;
  float v[4], s = 0.f, s2 = 0.f;
#pragma unroll
  for (int i = 0; i < 4; i++) {
    v[i] = row[lane + 32 * i];
    s += v[i];
    s2 += v[i] * v[i];
  }
#pragma unroll
  for (int off = 1; off < 32; off <<= 1) {
    s  += __shfl_xor(s, off, 32);
    s2 += __shfl_xor(s2, off, 32);
  }
  float mu  = s * (1.0f / 128.0f);
  float var = s2 * (1.0f / 128.0f) - mu * mu;
  float inv = rsqrtf(var + 1e-5f);
#pragma unroll
  for (int i = 0; i < 4; i++) {
    int col = lane + 32 * i;
    xn[(size_t)wave * D_DIM + col] = (v[i] - mu) * inv * g[col] + b[col];
  }
}

__global__ void zero_kernel(float* __restrict__ p, size_t n) {
  size_t i = (size_t)blockIdx.x * blockDim.x + threadIdx.x;
  if (i < n) p[i] = 0.0f;
}

// ---------------------------------------------------------------------------
// Edge kernel: per wave, one 16-edge tile through gate MLP + radial MLP,
// then fused gather * w_rad * edge_w -> atomic scatter-add.
// LDS: packed edge weights (40960 halves, TDM-staged) + per-wave staging.
// ---------------------------------------------------------------------------
#define WAVES_E 8
__global__ __launch_bounds__(256) void edge_kernel(
    const float* __restrict__ xn, const int* __restrict__ esrc, const int* __restrict__ edst,
    const float* __restrict__ erbf, const float* __restrict__ elen,
    const _Float16* __restrict__ packedE,
    const float* __restrict__ b1, const float* __restrict__ b2, const float* __restrict__ b3,
    const float* __restrict__ bg1, const float* __restrict__ g2w, const float* __restrict__ bg2p,
    float* __restrict__ agg, float* __restrict__ normbuf, int numTiles) {
  extern __shared__ _Float16 sm[];
  _Float16* sW1 = sm;                 // 4096 halves  (K=32)
  _Float16* sW2 = sm + 4096;          // 16384 halves (K=128)
  _Float16* sW3 = sm + 20480;         // 16384 halves (K=128)
  _Float16* sG1 = sm + 36864;         // 4096 halves  (K=32)
  _Float16* sStage = sm + 40960;      // 8 waves * 2048 halves

  const int lane = threadIdx.x & 31;
  const int wave = threadIdx.x >> 5;

#if HAVE_TDM
  if (wave == 0) {  // one wave drives the DMA: 40960 halves = 20480 dwords = 4096 x 5
    unsigned lds_off = (unsigned)((size_t)(void*)sm & 0xffffffffu);
    tdm_load_2d_dwords(packedE, lds_off, 4096u, 5u);
    __builtin_amdgcn_s_wait_tensorcnt(0);
  }
#else
  {
    const uint4* s = (const uint4*)packedE;
    uint4* d = (uint4*)sm;
    for (int i = threadIdx.x; i < 5120; i += 256) d[i] = s[i];
  }
#endif
  __syncthreads();

  const int hi = lane >> 4;      // which K-half group this lane holds
  const int n16 = lane & 15;     // column within 16-wide tile / A-row index
  _Float16* stage = sStage + wave * 2048;
  const float bg2 = bg2p[0];

  for (int tile = blockIdx.x * WAVES_E + wave; tile < numTiles; tile += gridDim.x * WAVES_E) {
    const int tbase = tile * 16;

    // --- rbf A fragment: row m = n16, K groups {hi*8..+7} and {hi*8+16..+7} ---
    const float* rrow = erbf + (size_t)(tbase + n16) * 32;
    const int k0 = hi * 8;
    v16h a_rbf = pack16(*(const float4*)(rrow + k0),      *(const float4*)(rrow + k0 + 4),
                        *(const float4*)(rrow + k0 + 16), *(const float4*)(rrow + k0 + 20), 1.0f);

    {  // prefetch next tile's rbf row for this wave
      int nt = tile + gridDim.x * WAVES_E;
      if (nt < numTiles) __builtin_prefetch(erbf + (size_t)(nt * 16 + n16) * 32 + k0, 0, 1);
    }

    // --- per-row metadata (C-layout rows m = j + hi*8) ---
    int srcv[8], dstv[8];
    float ew[8];
#pragma unroll
    for (int j = 0; j < 8; j++) {
      int e = tbase + j + hi * 8;
      srcv[j] = esrc[e];
      dstv[j] = edst[e];
    }

    // --- edge gate MLP: g = silu(rbf@G1+bg1); pre = g . G2 + bg2 ---
    float part[8];
#pragma unroll
    for (int j = 0; j < 8; j++) part[j] = 0.0f;
#pragma unroll
    for (int c = 0; c < 8; c++) {
      int col = c * 16 + n16;
      float bb = bg1[col];
      v8f acc;
#pragma unroll
      for (int t = 0; t < 8; t++) acc[t] = bb;
      v16h bf = *(const v16h*)(sG1 + (c * 32 + lane) * 16);
      acc = wmma_f16(a_rbf, bf, acc);
      float gw = g2w[col];
#pragma unroll
      for (int j = 0; j < 8; j++) part[j] += siluf(acc[j]) * gw;
    }
#pragma unroll
    for (int j = 0; j < 8; j++) part[j] = gsum16(part[j]);
#pragma unroll
    for (int j = 0; j < 8; j++) {
      float len = elen[tbase + j + hi * 8];
      float cut = (len <= CUTOFF_F) ? 0.5f * (__cosf(PI_OVER_CUTOFF * len) + 1.0f) : 0.0f;
      ew[j] = cut * sigmf(part[j] + bg2);
    }
    if (n16 == 0) {  // one lane per row-group scatters edge_w
#pragma unroll
      for (int j = 0; j < 8; j++) atomicAdd(normbuf + dstv[j], ew[j]);
    }

    // --- radial layer 1: h1 = silu(rbf@W1+b1), stage C-layout -> row-major LDS ---
#pragma unroll
    for (int c = 0; c < 8; c++) {
      int col = c * 16 + n16;
      float bb = b1[col];
      v8f acc;
#pragma unroll
      for (int t = 0; t < 8; t++) acc[t] = bb;
      v16h bf = *(const v16h*)(sW1 + (c * 32 + lane) * 16);
      acc = wmma_f16(a_rbf, bf, acc);
#pragma unroll
      for (int j = 0; j < 8; j++)
        stage[(j + hi * 8) * D_DIM + col] = (_Float16)siluf(acc[j]);
    }
    asm volatile("s_wait_dscnt 0" ::: "memory");
    v16h ah[4];
#pragma unroll
    for (int kc = 0; kc < 4; kc++) ah[kc] = load_a_stage(stage, kc, n16, hi);

    // --- radial layer 2: h2 = silu(h1@W2+b2); batch B frags, then WMMA chain ---
#pragma unroll
    for (int c = 0; c < 8; c++) {
      int col = c * 16 + n16;
      v16h bf[4];
#pragma unroll
      for (int kc = 0; kc < 4; kc++)
        bf[kc] = *(const v16h*)(sW2 + ((kc * 8 + c) * 32 + lane) * 16);
      float bb = b2[col];
      v8f acc;
#pragma unroll
      for (int t = 0; t < 8; t++) acc[t] = bb;
#pragma unroll
      for (int kc = 0; kc < 4; kc++) acc = wmma_f16(ah[kc], bf[kc], acc);
#pragma unroll
      for (int j = 0; j < 8; j++)
        stage[(j + hi * 8) * D_DIM + col] = (_Float16)siluf(acc[j]);
    }
    asm volatile("s_wait_dscnt 0" ::: "memory");
#pragma unroll
    for (int kc = 0; kc < 4; kc++) ah[kc] = load_a_stage(stage, kc, n16, hi);

    // --- radial layer 3 + fused gather / gate / atomic scatter ---
#pragma unroll
    for (int c = 0; c < 8; c++) {
      int col = c * 16 + n16;
      v16h bf[4];
#pragma unroll
      for (int kc = 0; kc < 4; kc++)
        bf[kc] = *(const v16h*)(sW3 + ((kc * 8 + c) * 32 + lane) * 16);
      float bb = b3[col];
      v8f acc;
#pragma unroll
      for (int t = 0; t < 8; t++) acc[t] = bb;
#pragma unroll
      for (int kc = 0; kc < 4; kc++) acc = wmma_f16(ah[kc], bf[kc], acc);
#pragma unroll
      for (int j = 0; j < 8; j++) {
        float xv = xn[(size_t)srcv[j] * D_DIM + col];   // L2-resident gather
        float mv = xv * acc[j] * ew[j];
        atomicAdd(agg + (size_t)dstv[j] * D_DIM + col, mv);
      }
    }
  }
}

// ---------------------------------------------------------------------------
// Node kernel: agg/norm -> message MLP -> self/update linears -> residual.
// LDS: packed node weights (65536 halves, TDM-staged) + per-wave staging.
// ---------------------------------------------------------------------------
#define WAVES_N 4
__global__ __launch_bounds__(128) void node_kernel(
    const float* __restrict__ xn, const float* __restrict__ x,
    const float* __restrict__ agg, const float* __restrict__ normbuf,
    const _Float16* __restrict__ packedN,
    const float* __restrict__ mmb1, const float* __restrict__ mmb2,
    const float* __restrict__ slb, const float* __restrict__ ulb,
    const float* __restrict__ rsp, float* __restrict__ out, int numTiles) {
  extern __shared__ _Float16 sm[];
  _Float16* sMM1 = sm;                // 16384
  _Float16* sMM2 = sm + 16384;        // 16384
  _Float16* sSL  = sm + 32768;        // 16384
  _Float16* sUL  = sm + 49152;        // 16384
  _Float16* sStage = sm + 65536;      // 4 waves * 2048

  const int lane = threadIdx.x & 31;
  const int wave = threadIdx.x >> 5;

#if HAVE_TDM
  if (wave == 0) {  // 65536 halves = 32768 dwords = 8192 x 4
    unsigned lds_off = (unsigned)((size_t)(void*)sm & 0xffffffffu);
    tdm_load_2d_dwords(packedN, lds_off, 8192u, 4u);
    __builtin_amdgcn_s_wait_tensorcnt(0);
  }
#else
  {
    const uint4* s = (const uint4*)packedN;
    uint4* d = (uint4*)sm;
    for (int i = threadIdx.x; i < 8192; i += 128) d[i] = s[i];
  }
#endif
  __syncthreads();

  const int hi = lane >> 4;
  const int n16 = lane & 15;
  _Float16* stage = sStage + wave * 2048;
  const float rs = rsp[0];

  for (int tile = blockIdx.x * WAVES_N + wave; tile < numTiles; tile += gridDim.x * WAVES_N) {
    const int tbase = tile * 16;

    // --- normalized-agg A fragments (4 K-chunks), row m = n16 ---
    v16h ah[4];
    {
      int node = tbase + n16;
      float nv = normbuf[node];
      float inv = __builtin_amdgcn_rcpf(nv > 1e-8f ? nv : 1e-8f);
      const float* row = agg + (size_t)node * D_DIM;
#pragma unroll
      for (int kc = 0; kc < 4; kc++) {
        int k0 = kc * 32 + hi * 8;
        ah[kc] = pack16(*(const float4*)(row + k0),      *(const float4*)(row + k0 + 4),
                        *(const float4*)(row + k0 + 16), *(const float4*)(row + k0 + 20), inv);
      }
    }

    // --- t1 = silu(aggn @ mm_w1 + mm_b1) ---
#pragma unroll
    for (int c = 0; c < 8; c++) {
      int col = c * 16 + n16;
      v16h bf[4];
#pragma unroll
      for (int kc = 0; kc < 4; kc++)
        bf[kc] = *(const v16h*)(sMM1 + ((kc * 8 + c) * 32 + lane) * 16);
      float bb = mmb1[col];
      v8f acc;
#pragma unroll
      for (int t = 0; t < 8; t++) acc[t] = bb;
#pragma unroll
      for (int kc = 0; kc < 4; kc++) acc = wmma_f16(ah[kc], bf[kc], acc);
#pragma unroll
      for (int j = 0; j < 8; j++)
        stage[(j + hi * 8) * D_DIM + col] = (_Float16)siluf(acc[j]);
    }
    asm volatile("s_wait_dscnt 0" ::: "memory");
#pragma unroll
    for (int kc = 0; kc < 4; kc++) ah[kc] = load_a_stage(stage, kc, n16, hi);

    // --- agg2 = t1 @ mm_w2 + mm_b2 (staged f16 for reuse as A) ---
#pragma unroll
    for (int c = 0; c < 8; c++) {
      int col = c * 16 + n16;
      v16h bf[4];
#pragma unroll
      for (int kc = 0; kc < 4; kc++)
        bf[kc] = *(const v16h*)(sMM2 + ((kc * 8 + c) * 32 + lane) * 16);
      float bb = mmb2[col];
      v8f acc;
#pragma unroll
      for (int t = 0; t < 8; t++) acc[t] = bb;
#pragma unroll
      for (int kc = 0; kc < 4; kc++) acc = wmma_f16(ah[kc], bf[kc], acc);
#pragma unroll
      for (int j = 0; j < 8; j++)
        stage[(j + hi * 8) * D_DIM + col] = (_Float16)acc[j];
    }
    asm volatile("s_wait_dscnt 0" ::: "memory");
    v16h ag2[4];
#pragma unroll
    for (int kc = 0; kc < 4; kc++) ag2[kc] = load_a_stage(stage, kc, n16, hi);

    // --- x_norm A fragments from global (L2-resident) ---
    v16h axn[4];
    {
      const float* row = xn + (size_t)(tbase + n16) * D_DIM;
#pragma unroll
      for (int kc = 0; kc < 4; kc++) {
        int k0 = kc * 32 + hi * 8;
        axn[kc] = pack16(*(const float4*)(row + k0),      *(const float4*)(row + k0 + 4),
                         *(const float4*)(row + k0 + 16), *(const float4*)(row + k0 + 20), 1.0f);
      }
    }

    // --- out = x + rs * (xn@sl_w + sl_b + agg2@ul_w + ul_b) ---
#pragma unroll
    for (int c = 0; c < 8; c++) {
      int col = c * 16 + n16;
      v16h bfs[4], bfu[4];
#pragma unroll
      for (int kc = 0; kc < 4; kc++) {
        bfs[kc] = *(const v16h*)(sSL + ((kc * 8 + c) * 32 + lane) * 16);
        bfu[kc] = *(const v16h*)(sUL + ((kc * 8 + c) * 32 + lane) * 16);
      }
      float bb = slb[col] + ulb[col];
      v8f acc;
#pragma unroll
      for (int t = 0; t < 8; t++) acc[t] = bb;
#pragma unroll
      for (int kc = 0; kc < 4; kc++) {
        acc = wmma_f16(axn[kc], bfs[kc], acc);
        acc = wmma_f16(ag2[kc], bfu[kc], acc);
      }
#pragma unroll
      for (int j = 0; j < 8; j++) {
        size_t idx = (size_t)(tbase + j + hi * 8) * D_DIM + col;
        out[idx] = x[idx] + rs * acc[j];
      }
    }
  }
}

// ---------------------------------------------------------------------------
extern "C" void kernel_launch(void* const* d_in, const int* in_sizes, int n_in,
                              void* d_out, int out_size, void* d_ws, size_t ws_size,
                              hipStream_t stream) {
  const float* x     = (const float*)d_in[0];
  const int*   esrc  = (const int*)d_in[1];
  const int*   edst  = (const int*)d_in[2];
  // d_in[3] edge_sh: unused by the reference
  const float* erbf  = (const float*)d_in[4];
  const float* elen  = (const float*)d_in[5];
  const float* lng   = (const float*)d_in[6];
  const float* lnb   = (const float*)d_in[7];
  const float* wm_w1 = (const float*)d_in[8];
  const float* wm_b1 = (const float*)d_in[9];
  const float* wm_w2 = (const float*)d_in[10];
  const float* wm_b2 = (const float*)d_in[11];
  const float* wm_w3 = (const float*)d_in[12];
  const float* wm_b3 = (const float*)d_in[13];
  const float* eg_w1 = (const float*)d_in[14];
  const float* eg_b1 = (const float*)d_in[15];
  const float* eg_w2 = (const float*)d_in[16];
  const float* eg_b2 = (const float*)d_in[17];
  const float* mm_w1 = (const float*)d_in[18];
  const float* mm_b1 = (const float*)d_in[19];
  const float* mm_w2 = (const float*)d_in[20];
  const float* mm_b2 = (const float*)d_in[21];
  const float* sl_w  = (const float*)d_in[22];
  const float* sl_b  = (const float*)d_in[23];
  const float* ul_w  = (const float*)d_in[24];
  const float* ul_b  = (const float*)d_in[25];
  const float* rsp   = (const float*)d_in[26];

  const int N = in_sizes[0] / D_DIM;
  const int E = in_sizes[1];

  // Workspace layout (fp32 then f16):
  float* xnorm   = (float*)d_ws;                        // N*128
  float* agg     = xnorm + (size_t)N * D_DIM;           // N*128
  float* normbuf = agg + (size_t)N * D_DIM;             // N
  _Float16* packedE = (_Float16*)(normbuf + N);         // 40960 halves
  _Float16* packedN = packedE + 40960;                  // 65536 halves

  // 1) repack weights into WMMA B-fragment order (f16)
  pack_weights<<<16, 256, 0, stream>>>(wm_w1, packedE, 32);
  pack_weights<<<64, 256, 0, stream>>>(wm_w2, packedE + 4096, 128);
  pack_weights<<<64, 256, 0, stream>>>(wm_w3, packedE + 20480, 128);
  pack_weights<<<16, 256, 0, stream>>>(eg_w1, packedE + 36864, 32);
  pack_weights<<<64, 256, 0, stream>>>(mm_w1, packedN, 128);
  pack_weights<<<64, 256, 0, stream>>>(mm_w2, packedN + 16384, 128);
  pack_weights<<<64, 256, 0, stream>>>(sl_w,  packedN + 32768, 128);
  pack_weights<<<64, 256, 0, stream>>>(ul_w,  packedN + 49152, 128);

  // 2) LayerNorm
  ln_kernel<<<(N + 7) / 8, 256, 0, stream>>>(x, lng, lnb, xnorm, N);

  // 3) zero accumulators (agg + normbuf are contiguous)
  size_t zn = (size_t)N * D_DIM + N;
  zero_kernel<<<(int)((zn + 255) / 256), 256, 0, stream>>>(agg, zn);

  // 4) fused edge pipeline (WMMA); 1024 blocks x 8 waves grid-stride over tiles
  int numTilesE = E / 16;
  edge_kernel<<<1024, 256, 114688, stream>>>(xnorm, esrc, edst, erbf, elen, packedE,
                                             wm_b1, wm_b2, wm_b3, eg_b1, eg_w2, eg_b2,
                                             agg, normbuf, numTilesE);

  // 5) node update (WMMA) + residual write
  int numTilesN = N / 16;
  node_kernel<<<(numTilesN + WAVES_N - 1) / WAVES_N, 128, 147456, stream>>>(
      xnorm, x, agg, normbuf, packedN, mm_b1, mm_b2, sl_b, ul_b, rsp,
      (float*)d_out, numTilesN);
}
